// NNG_20486994002510
// MI455X (gfx1250) — matrix operand order repaired and checked by reading, not compiled, not run on hardware
//
#include <hip/hip_runtime.h>
#include <math.h>

#define H 128
#define E_EDGES 393216
#define G_NODES 131072

typedef __bf16 bf16_t;
typedef __attribute__((ext_vector_type(16))) __bf16 v16bf;
typedef __attribute__((ext_vector_type(8)))  __bf16 v8bf;
typedef __attribute__((ext_vector_type(8)))  float  v8f;
typedef __attribute__((ext_vector_type(4)))  float  v4f;

__device__ __forceinline__ v8f wmma_bf16(v16bf a, v16bf b, v8f c) {
    return __builtin_amdgcn_wmma_f32_16x16x32_bf16(
        /*neg_a=*/false, a, /*neg_b=*/false, b,
        /*c_mod=*/(short)0, c, /*reuse_a=*/false, /*reuse_b=*/false);
}

__device__ __forceinline__ float fast_silu(float x) {
    // x * sigmoid(x) with v_rcp_f32 instead of precise IEEE division
    return x * __builtin_amdgcn_rcpf(1.f + __expf(-x));
}

__device__ __forceinline__ void zero_acc(v8f acc[2][8]) {
#pragma unroll
    for (int mt = 0; mt < 2; ++mt)
#pragma unroll
        for (int nt = 0; nt < 8; ++nt)
#pragma unroll
            for (int i = 0; i < 8; ++i) acc[mt][nt][i] = 0.f;
}

// A staged row-major bf16 in LDS ([32][ldk], rows 16B-aligned); B pre-packed
// fragments in global memory: ((kt*8 + nt)*32 + lane)*16 + e.
// Two 16-row M tiles per wave: every B fragment feeds two WMMAs.
__device__ __forceinline__ void gemm_tile2(const bf16_t* aLds, int ldk, int kdim,
                                           const bf16_t* __restrict__ Bp,
                                           v8f acc[2][8], int lane) {
    const int m = lane & 15, h = lane >> 4;
    const int nkt = kdim >> 5;
    for (int kt = 0; kt < nkt; ++kt) {
        v16bf a[2];
#pragma unroll
        for (int mt = 0; mt < 2; ++mt) {
            const bf16_t* row = aLds + (mt * 16 + m) * ldk + h * 16 + kt * 32;
            v8bf lo = *(const v8bf*)(row);
            v8bf hi = *(const v8bf*)(row + 8);
#pragma unroll
            for (int i = 0; i < 8; ++i) { a[mt][i] = lo[i]; a[mt][i + 8] = hi[i]; }
        }
        const bf16_t* bbase = Bp + (size_t)kt * 4096 + lane * 16;
#pragma unroll
        for (int nt = 0; nt < 8; ++nt) {
            v16bf b = *(const v16bf*)(bbase + nt * 512);
            acc[0][nt] = wmma_bf16(a[0], b, acc[0][nt]);
            acc[1][nt] = wmma_bf16(a[1], b, acc[1][nt]);
        }
    }
}

// bias + SiLU, write bf16 C tiles back to LDS row-major at column coloff
__device__ __forceinline__ void bias_silu_store(const v8f acc[2][8],
                                                const float* __restrict__ bias,
                                                bf16_t* lds, int ldk, int coloff,
                                                int lane) {
    const int nlo = lane & 15, m0 = (lane >> 4) * 8;
#pragma unroll
    for (int nt = 0; nt < 8; ++nt) {
        float b = bias[nt * 16 + nlo];
#pragma unroll
        for (int mt = 0; mt < 2; ++mt)
#pragma unroll
            for (int i = 0; i < 8; ++i) {
                float s = fast_silu(acc[mt][nt][i] + b);
                lds[(mt * 16 + m0 + i) * ldk + coloff + nt * 16 + nlo] = (bf16_t)s;
            }
    }
}

__device__ __forceinline__ void store_c(const v8f acc[2][8], bf16_t* lds, int ldk,
                                        int coloff, int lane) {
    const int nlo = lane & 15, m0 = (lane >> 4) * 8;
#pragma unroll
    for (int nt = 0; nt < 8; ++nt)
#pragma unroll
        for (int mt = 0; mt < 2; ++mt)
#pragma unroll
            for (int i = 0; i < 8; ++i)
                lds[(mt * 16 + m0 + i) * ldk + coloff + nt * 16 + nlo] =
                    (bf16_t)acc[mt][nt][i];
}

// bias add + LayerNorm over N=128, in place on C fragments.
__device__ __forceinline__ void bias_ln(v8f acc[2][8],
                                        const float* __restrict__ bias,
                                        const float* __restrict__ g,
                                        const float* __restrict__ be, int lane) {
    const int nlo = lane & 15;
#pragma unroll
    for (int nt = 0; nt < 8; ++nt) {
        float b = bias[nt * 16 + nlo];
#pragma unroll
        for (int mt = 0; mt < 2; ++mt)
#pragma unroll
            for (int i = 0; i < 8; ++i) acc[mt][nt][i] += b;
    }
    float s[2][8], q[2][8];
#pragma unroll
    for (int mt = 0; mt < 2; ++mt)
#pragma unroll
        for (int i = 0; i < 8; ++i) { s[mt][i] = 0.f; q[mt][i] = 0.f; }
#pragma unroll
    for (int nt = 0; nt < 8; ++nt)
#pragma unroll
        for (int mt = 0; mt < 2; ++mt)
#pragma unroll
            for (int i = 0; i < 8; ++i) {
                float x = acc[mt][nt][i];
                s[mt][i] += x;
                q[mt][i] += x * x;
            }
#pragma unroll
    for (int off = 1; off < 16; off <<= 1) {
#pragma unroll
        for (int mt = 0; mt < 2; ++mt)
#pragma unroll
            for (int i = 0; i < 8; ++i) {
                s[mt][i] += __shfl_xor(s[mt][i], off, 32);
                q[mt][i] += __shfl_xor(q[mt][i], off, 32);
            }
    }
#pragma unroll
    for (int mt = 0; mt < 2; ++mt)
#pragma unroll
        for (int i = 0; i < 8; ++i) {
            float mu = s[mt][i] * (1.f / 128.f);
            float var = q[mt][i] * (1.f / 128.f) - mu * mu;
            s[mt][i] = mu;
            q[mt][i] = __builtin_amdgcn_rsqf(var + 1e-5f);
        }
#pragma unroll
    for (int nt = 0; nt < 8; ++nt) {
        float gg = g[nt * 16 + nlo], bb = be[nt * 16 + nlo];
#pragma unroll
        for (int mt = 0; mt < 2; ++mt)
#pragma unroll
            for (int i = 0; i < 8; ++i)
                acc[mt][nt][i] = (acc[mt][nt][i] - s[mt][i]) * q[mt][i] * gg + bb;
    }
}

// ---------------------------------------------------------------------------
__global__ void zero_kernel(float4* p, int n4) {
    int t = blockIdx.x * blockDim.x + threadIdx.x;
    if (t < n4) p[t] = make_float4(0.f, 0.f, 0.f, 0.f);
}

// Pack 9 weight matrices [K x 128] into WMMA B-fragment layout (bf16).
// Fragment element (kt, nt, lane, e): K = kt*32 + (lane/16)*16 + e,
// N = nt*16 + lane%16.  em_W1 zero-padded K: 4 -> 32.
__global__ void pack_kernel(const float* emW1, const float* emW2,
                            const float* edW1, const float* edW2, const float* edW3,
                            const float* nmW1, const float* nmW2,
                            const float* fpW1, const float* fpW2,
                            bf16_t* packed) {
    const int padK[9]  = {32, 128, 384, 128, 128, 256, 128, 128, 128};
    const int realK[9] = {4, 128, 384, 128, 128, 256, 128, 128, 128};
    const float* srcs[9] = {emW1, emW2, edW1, edW2, edW3, nmW1, nmW2, fpW1, fpW2};
    int t = blockIdx.x * blockDim.x + threadIdx.x;
    if (t >= 184320) return;
    int m = 0, off = 0;
    while (m < 9) {
        int sz = padK[m] * 128;
        if (t < off + sz) break;
        off += sz;
        ++m;
    }
    int u = t - off;
    int e = u & 15;
    int lanei = (u >> 4) & 31;
    int nt = (u >> 9) & 7;
    int kt = u >> 12;
    int k = kt * 32 + (lanei >> 4) * 16 + e;
    int n = nt * 16 + (lanei & 15);
    float v = (k < realK[m]) ? srcs[m][(size_t)k * 128 + n] : 0.f;
    packed[t] = (bf16_t)v;
}

// ---------------------------------------------------------------------------
// Edge pipeline: one wave per 32-edge double tile, 2 waves per block.
__global__ __launch_bounds__(64) void edge_kernel(
    const float* __restrict__ ef, const float* __restrict__ grid,
    const float* __restrict__ mesh, const int* __restrict__ srcI,
    const int* __restrict__ dstI,
    const float* __restrict__ em_b1, const float* __restrict__ em_b2,
    const float* __restrict__ em_g, const float* __restrict__ em_be,
    const float* __restrict__ ed_b1, const float* __restrict__ ed_b2,
    const float* __restrict__ ed_b3, const float* __restrict__ ed_g,
    const float* __restrict__ ed_be,
    const bf16_t* __restrict__ emW1p, const bf16_t* __restrict__ emW2p,
    const bf16_t* __restrict__ edW1p, const bf16_t* __restrict__ edW2p,
    const bf16_t* __restrict__ edW3p, float* __restrict__ agg) {
    constexpr int LDK = 392;  // 384 + 8 pad -> bank-conflict-free b128 rows
    __shared__ __align__(16) bf16_t stage[2][32 * LDK];
    __shared__ int dsts[2][32];
    const int wid = threadIdx.x >> 5;
    const int lane = threadIdx.x & 31;
    bf16_t* S = stage[wid];
    const int base = (blockIdx.x * 2 + wid) * 32;

    // --- stage zero-padded edge-feature A tile [32 x 32] at cols [0,32)
    {
        bf16_t* p = S + lane * LDK;
#pragma unroll
        for (int i = 0; i < 32; ++i) p[i] = (bf16_t)0.f;
        v4f v = *(const v4f*)(ef + (size_t)(base + lane) * 4);
        p[0] = (bf16_t)v[0]; p[1] = (bf16_t)v[1];
        p[2] = (bf16_t)v[2]; p[3] = (bf16_t)v[3];
        dsts[wid][lane] = dstI[base + lane];
    }
    __syncthreads();

    v8f acc[2][8];
    // --- edge embedding MLP: silu(ef@W1+b1)@W2+b2, then LN
    zero_acc(acc);
    gemm_tile2(S, LDK, 32, emW1p, acc, lane);
    bias_silu_store(acc, em_b1, S, LDK, 0, lane);
    __syncthreads();
    zero_acc(acc);
    gemm_tile2(S, LDK, 128, emW2p, acc, lane);
    bias_ln(acc, em_b2, em_g, em_be, lane);
    store_c(acc, S, LDK, 0, lane);  // emb -> cols [0,128)

    // --- gather mesh_feat[src] -> cols [128,256), grid_feat[dst] -> [256,384)
    {
        int si = srcI[base + lane];
        int di = dsts[wid][lane];
        const float* ms = mesh + (size_t)si * H;
        const float* gs = grid + (size_t)di * H;
        bf16_t* pm = S + lane * LDK + 128;
        bf16_t* pg = S + lane * LDK + 256;
#pragma unroll
        for (int j = 0; j < 32; ++j) {
            v4f a = *(const v4f*)(ms + j * 4);
            v4f b = *(const v4f*)(gs + j * 4);
            pm[j * 4 + 0] = (bf16_t)a[0]; pm[j * 4 + 1] = (bf16_t)a[1];
            pm[j * 4 + 2] = (bf16_t)a[2]; pm[j * 4 + 3] = (bf16_t)a[3];
            pg[j * 4 + 0] = (bf16_t)b[0]; pg[j * 4 + 1] = (bf16_t)b[1];
            pg[j * 4 + 2] = (bf16_t)b[2]; pg[j * 4 + 3] = (bf16_t)b[3];
        }
    }
    __syncthreads();

    // --- edge MLP: 3H -> H -> H -> H (+LN)
    zero_acc(acc);
    gemm_tile2(S, LDK, 384, edW1p, acc, lane);
    bias_silu_store(acc, ed_b1, S, LDK, 0, lane);  // h1 -> cols [0,128)
    __syncthreads();
    zero_acc(acc);
    gemm_tile2(S, LDK, 128, edW2p, acc, lane);
    bias_silu_store(acc, ed_b2, S, LDK, 128, lane);  // h2 -> cols [128,256)
    __syncthreads();
    zero_acc(acc);
    gemm_tile2(S + 128, LDK, 128, edW3p, acc, lane);
    bias_ln(acc, ed_b3, ed_g, ed_be, lane);

    // --- segment-sum scatter: agg[dst] += e
    {
        const int nlo = lane & 15, m0 = (lane >> 4) * 8;
#pragma unroll
        for (int mt = 0; mt < 2; ++mt)
#pragma unroll
            for (int nt = 0; nt < 8; ++nt)
#pragma unroll
                for (int i = 0; i < 8; ++i) {
                    int d = dsts[wid][mt * 16 + m0 + i];
                    atomicAdd(agg + (size_t)d * H + nt * 16 + nlo, acc[mt][nt][i]);
                }
    }
}

// ---------------------------------------------------------------------------
// Node pipeline: one wave per 32-node double tile, 2 waves per block.
__global__ __launch_bounds__(64) void node_kernel(
    const float* __restrict__ grid, const float* __restrict__ agg,
    const float* __restrict__ nm_b1, const float* __restrict__ nm_b2,
    const float* __restrict__ nm_g, const float* __restrict__ nm_be,
    const float* __restrict__ fp_b1, const float* __restrict__ fp_b2,
    const float* __restrict__ fp_g, const float* __restrict__ fp_be,
    const bf16_t* __restrict__ nmW1p, const bf16_t* __restrict__ nmW2p,
    const bf16_t* __restrict__ fpW1p, const bf16_t* __restrict__ fpW2p,
    float* __restrict__ out) {
    constexpr int LDK = 264;  // 256 + 8 pad
    __shared__ __align__(16) bf16_t stage[2][32 * LDK];
    const int wid = threadIdx.x >> 5;
    const int lane = threadIdx.x & 31;
    bf16_t* S = stage[wid];
    const int base = (blockIdx.x * 2 + wid) * 32;

    // stage [agg | grid] -> cols [0,128) | [128,256)
    {
        const float* pa = agg + (size_t)(base + lane) * H;
        const float* pg = grid + (size_t)(base + lane) * H;
        bf16_t* qa = S + lane * LDK;
        bf16_t* qg = S + lane * LDK + 128;
#pragma unroll
        for (int j = 0; j < 32; ++j) {
            v4f a = *(const v4f*)(pa + j * 4);
            v4f b = *(const v4f*)(pg + j * 4);
            qa[j * 4 + 0] = (bf16_t)a[0]; qa[j * 4 + 1] = (bf16_t)a[1];
            qa[j * 4 + 2] = (bf16_t)a[2]; qa[j * 4 + 3] = (bf16_t)a[3];
            qg[j * 4 + 0] = (bf16_t)b[0]; qg[j * 4 + 1] = (bf16_t)b[1];
            qg[j * 4 + 2] = (bf16_t)b[2]; qg[j * 4 + 3] = (bf16_t)b[3];
        }
    }
    __syncthreads();

    v8f acc[2][8];
    const int nlo = lane & 15, m0 = (lane >> 4) * 8;

    // node_mlp: silu(cat2@W1+b1)@W2+b2, LN, + residual grid_feat
    zero_acc(acc);
    gemm_tile2(S, LDK, 256, nmW1p, acc, lane);
    bias_silu_store(acc, nm_b1, S, LDK, 0, lane);  // h -> cols [0,128)
    __syncthreads();
    zero_acc(acc);
    gemm_tile2(S, LDK, 128, nmW2p, acc, lane);
    bias_ln(acc, nm_b2, nm_g, nm_be, lane);
#pragma unroll
    for (int mt = 0; mt < 2; ++mt)
#pragma unroll
        for (int nt = 0; nt < 8; ++nt)
#pragma unroll
            for (int i = 0; i < 8; ++i)
                acc[mt][nt][i] +=
                    grid[(size_t)(base + mt * 16 + m0 + i) * H + nt * 16 + nlo];
    store_c(acc, S, LDK, 128, lane);  // decoded -> cols [128,256)
    __syncthreads();

    // frame projection: silu(decoded@W1+b1)@W2+b2, LN
    zero_acc(acc);
    gemm_tile2(S + 128, LDK, 128, fpW1p, acc, lane);
    bias_silu_store(acc, fp_b1, S, LDK, 0, lane);
    __syncthreads();
    zero_acc(acc);
    gemm_tile2(S, LDK, 128, fpW2p, acc, lane);
    bias_ln(acc, fp_b2, fp_g, fp_be, lane);

#pragma unroll
    for (int mt = 0; mt < 2; ++mt)
#pragma unroll
        for (int nt = 0; nt < 8; ++nt)
#pragma unroll
            for (int i = 0; i < 8; ++i)
                out[(size_t)(base + mt * 16 + m0 + i) * H + nt * 16 + nlo] =
                    acc[mt][nt][i];
}

// ---------------------------------------------------------------------------
extern "C" void kernel_launch(void* const* d_in, const int* in_sizes, int n_in,
                              void* d_out, int out_size, void* d_ws, size_t ws_size,
                              hipStream_t stream) {
    const float* ef    = (const float*)d_in[0];
    const float* grid  = (const float*)d_in[1];
    const float* mesh  = (const float*)d_in[2];
    const int*   srcI  = (const int*)d_in[3];
    const int*   dstI  = (const int*)d_in[4];
    // d_in[5] = num_dst_nodes (compile-time G)
    const float* emW1 = (const float*)d_in[6];
    const float* em_b1 = (const float*)d_in[7];
    const float* emW2 = (const float*)d_in[8];
    const float* em_b2 = (const float*)d_in[9];
    const float* em_g = (const float*)d_in[10];
    const float* em_be = (const float*)d_in[11];
    const float* edW1 = (const float*)d_in[12];
    const float* ed_b1 = (const float*)d_in[13];
    const float* edW2 = (const float*)d_in[14];
    const float* ed_b2 = (const float*)d_in[15];
    const float* edW3 = (const float*)d_in[16];
    const float* ed_b3 = (const float*)d_in[17];
    const float* ed_g = (const float*)d_in[18];
    const float* ed_be = (const float*)d_in[19];
    const float* nmW1 = (const float*)d_in[20];
    const float* nm_b1 = (const float*)d_in[21];
    const float* nmW2 = (const float*)d_in[22];
    const float* nm_b2 = (const float*)d_in[23];
    const float* nm_g = (const float*)d_in[24];
    const float* nm_be = (const float*)d_in[25];
    const float* fpW1 = (const float*)d_in[26];
    const float* fp_b1 = (const float*)d_in[27];
    const float* fpW2 = (const float*)d_in[28];
    const float* fp_b2 = (const float*)d_in[29];
    const float* fp_g = (const float*)d_in[30];
    const float* fp_be = (const float*)d_in[31];

    float* agg = (float*)d_ws;
    bf16_t* packed = (bf16_t*)((char*)d_ws + (size_t)G_NODES * H * sizeof(float));
    // packed-weight offsets (bf16 elements)
    bf16_t* emW1p = packed + 0;
    bf16_t* emW2p = packed + 4096;
    bf16_t* edW1p = packed + 20480;
    bf16_t* edW2p = packed + 69632;
    bf16_t* edW3p = packed + 86016;
    bf16_t* nmW1p = packed + 102400;
    bf16_t* nmW2p = packed + 135168;
    bf16_t* fpW1p = packed + 151552;
    bf16_t* fpW2p = packed + 167936;

    zero_kernel<<<(G_NODES * H / 4 + 255) / 256, 256, 0, stream>>>((float4*)agg,
                                                                   G_NODES * H / 4);
    pack_kernel<<<(184320 + 255) / 256, 256, 0, stream>>>(
        emW1, emW2, edW1, edW2, edW3, nmW1, nmW2, fpW1, fpW2, packed);
    edge_kernel<<<E_EDGES / 64, 64, 0, stream>>>(
        ef, grid, mesh, srcI, dstI, em_b1, em_b2, em_g, em_be, ed_b1, ed_b2, ed_b3,
        ed_g, ed_be, emW1p, emW2p, edW1p, edW2p, edW3p, agg);
    node_kernel<<<G_NODES / 64, 64, 0, stream>>>(
        grid, agg, nm_b1, nm_b2, nm_g, nm_be, fp_b1, fp_b2, fp_g, fp_be, nmW1p,
        nmW2p, fpW1p, fpW2p, (float*)d_out);
}
